// MultiHeadAttention_85787676770904
// MI455X (gfx1250) — matrix operand order repaired and checked by reading, not compiled, not run on hardware
//
#include <hip/hip_runtime.h>
#include <hip/hip_bf16.h>

// MI455X (gfx1250) MHA forward.
// Memory-bound on the 537MB attn-probability write -> compute in f16 WMMA
// (f32 accumulate), stage the whole 16x2048 score strip in the 320KB WGP LDS
// so scores are computed exactly once.

typedef __attribute__((ext_vector_type(16))) _Float16 v16h;
typedef __attribute__((ext_vector_type(8)))  _Float16 v8h;
typedef __attribute__((ext_vector_type(8)))  float    v8f;

namespace {
constexpr int Bq = 2;
constexpr int S  = 2048;
constexpr int D  = 1024;
constexpr int H  = 16;
constexpr int DK = 64;
constexpr int SP = 2052;   // padded LDS row stride (floats): bank-conflict-free column reads
}

// 16x32 f16 A fragment: lane m=l&15; K = 16*(l>>4) + 0..15 contiguous
__device__ __forceinline__ v16h load_a_frag(const _Float16* rowp) {
  return *(const v16h*)(rowp);
}

// 32x16 f16 B fragment from BT (N x K row-major): lane n=l&15; K = 16*(l>>4) + 0..15 contiguous
__device__ __forceinline__ v16h load_b_frag(const _Float16* rowp) {
  return *(const v16h*)(rowp);
}

// ---------------------------------------------------------------------------
__global__ void cvt_f32_to_f16(const float* __restrict__ src, _Float16* __restrict__ dst, int n) {
  int i = blockIdx.x * blockDim.x + threadIdx.x;
  if (i < n) dst[i] = (_Float16)src[i];
}

// W (D x D f32, row-major) -> WT (D x D f16), WT[n*D + k] = W[k*D + n]
__global__ void wtrans_f16(const float* __restrict__ W, _Float16* __restrict__ WT) {
  int idx = blockIdx.x * blockDim.x + threadIdx.x;
  if (idx >= D * D) return;
  int n = idx >> 10;
  int k = idx & (D - 1);
  WT[idx] = (_Float16)W[(size_t)k * D + n];
}

// VT[((b*H+h)*DK + d)*S + s] = Vh[(b*S+s)*D + h*DK + d]
__global__ void vtrans_f16(const _Float16* __restrict__ Vh, _Float16* __restrict__ VT) {
  int idx = blockIdx.x * blockDim.x + threadIdx.x;
  if (idx >= Bq * H * DK * S) return;
  int s = idx & (S - 1);
  int r = idx >> 11;
  int d = r & (DK - 1);
  r >>= 6;
  int h = r & (H - 1);
  int b = r >> 4;
  VT[idx] = Vh[((size_t)b * S + s) * D + h * DK + d];
}

// ---------------------------------------------------------------------------
// C = A(MxK f16) @ B, with B supplied transposed as BT (NxK f16, row-major).
// Each wave computes a 16x64 C tile (4 WMMA column tiles), K-loop step 32.
// If Cf != null: Cf[m,n] = acc + resid[m,n] (f32). Else Ch[m,n] = (f16)acc.
__global__ void gemm_wmma_f16(const _Float16* __restrict__ A,
                              const _Float16* __restrict__ BT,
                              _Float16* __restrict__ Ch,
                              float* __restrict__ Cf,
                              const float* __restrict__ resid,
                              int M, int N, int K,
                              int lda, int ldb, int ldc) {
  const int gtid = blockIdx.x * blockDim.x + threadIdx.x;
  const int wave = gtid >> 5;
  const int lane = threadIdx.x & 31;
  const int nw = N >> 6;
  const int m0 = (wave / nw) << 4;
  const int n0 = (wave % nw) << 6;
  if (m0 >= M) return;                       // wave-uniform; grids sized exactly anyway
  const int mr   = lane & 15;
  const int kb8  = (lane >> 4) << 3;
  const int kb16 = (lane >> 4) << 4;

  v8f a0 = {}, a1 = {}, a2 = {}, a3 = {};
  const _Float16* arow  = A  + (size_t)(m0 + mr) * lda + kb16;
  const _Float16* brow0 = BT + (size_t)(n0 +  0 + mr) * ldb + kb16;
  const _Float16* brow1 = BT + (size_t)(n0 + 16 + mr) * ldb + kb16;
  const _Float16* brow2 = BT + (size_t)(n0 + 32 + mr) * ldb + kb16;
  const _Float16* brow3 = BT + (size_t)(n0 + 48 + mr) * ldb + kb16;

  for (int k0 = 0; k0 < K; k0 += 32) {
    v16h a = load_a_frag(arow + k0);
    a0 = __builtin_amdgcn_wmma_f32_16x16x32_f16(false, a, false, load_b_frag(brow0 + k0), (short)0, a0, false, false);
    a1 = __builtin_amdgcn_wmma_f32_16x16x32_f16(false, a, false, load_b_frag(brow1 + k0), (short)0, a1, false, false);
    a2 = __builtin_amdgcn_wmma_f32_16x16x32_f16(false, a, false, load_b_frag(brow2 + k0), (short)0, a2, false, false);
    a3 = __builtin_amdgcn_wmma_f32_16x16x32_f16(false, a, false, load_b_frag(brow3 + k0), (short)0, a3, false, false);
  }

  const int il = (lane >> 4) << 3;           // C/D layout: lane l, vgpr r -> m = r + 8*(l>=16)
  if (Cf) {
#pragma unroll
    for (int r = 0; r < 8; ++r) {
      const size_t row = (size_t)(m0 + il + r) * ldc;
      Cf[row + n0 +  0 + mr] = a0[r] + resid[row + n0 +  0 + mr];
      Cf[row + n0 + 16 + mr] = a1[r] + resid[row + n0 + 16 + mr];
      Cf[row + n0 + 32 + mr] = a2[r] + resid[row + n0 + 32 + mr];
      Cf[row + n0 + 48 + mr] = a3[r] + resid[row + n0 + 48 + mr];
    }
  } else {
#pragma unroll
    for (int r = 0; r < 8; ++r) {
      const size_t row = (size_t)(m0 + il + r) * ldc;
      Ch[row + n0 +  0 + mr] = (_Float16)a0[r];
      Ch[row + n0 + 16 + mr] = (_Float16)a1[r];
      Ch[row + n0 + 32 + mr] = (_Float16)a2[r];
      Ch[row + n0 + 48 + mr] = (_Float16)a3[r];
    }
  }
}

// ---------------------------------------------------------------------------
// One workgroup (8 waves) per (b, h, 16-row i-tile).
// Pass 1: scores = (K_head @ Q_head^T)/8 with mask -> LDS (16 x S, f32).
// Softmax over j in LDS. Write normalized attn to d_out. P@V via WMMA.
__global__ void attn_kernel(const _Float16* __restrict__ Qh,
                            const _Float16* __restrict__ Kh,
                            const _Float16* __restrict__ VT,
                            const int* __restrict__ mask,
                            float* __restrict__ attnOut,
                            _Float16* __restrict__ AO) {
  extern __shared__ float smem[];
  float* sS   = smem;            // 16 * SP scores/probs
  float* sRed = sS + 16 * SP;    // 256 reduce scratch
  float* sRow = sRed + 256;      // [0..15] row max, [16..31] 1/rowsum
  float* sOp  = sRow + 32;       // 8 waves * 16*64 partial O

  const int tid  = threadIdx.x;
  const int lane = tid & 31;
  const int wv   = tid >> 5;
  const int blk  = blockIdx.x;
  const int it = blk & 127;
  const int h  = (blk >> 7) & 15;
  const int b  = blk >> 11;
  const int i0 = it * 16;
  const int mr   = lane & 15;
  const int kb8  = (lane >> 4) << 3;
  const int kb16 = (lane >> 4) << 4;
  const int il   = (lane >> 4) << 3;

  const size_t headOff = (size_t)b * S * D + (size_t)h * DK;

  // ---- Pass 1: masked, scaled scores into LDS ----
  for (int t = wv; t < S / 16; t += 8) {
    const int j0 = t * 16;
    v8f acc = {};
#pragma unroll
    for (int kk = 0; kk < DK; kk += 32) {
      v16h a  = load_a_frag(Kh + headOff + (size_t)(i0 + mr) * D + kk + kb16);
      v16h bf = load_b_frag(Qh + headOff + (size_t)(j0 + mr) * D + kk + kb16);
      acc = __builtin_amdgcn_wmma_f32_16x16x32_f16(false, a, false, bf, (short)0, acc, false, false);
    }
    const int j = j0 + mr;
    const size_t mBase = ((size_t)b * S + i0) * S + j;
#pragma unroll
    for (int r = 0; r < 8; ++r) {
      const int i = il + r;
      const int mv = mask[mBase + (size_t)i * S];
      const float sc = acc[r] * 0.125f;               // /TEMP
      sS[i * SP + j] = mv ? sc : -1e-15f;             // MIN_VAL
    }
  }
  __syncthreads();

  // ---- softmax over j: max, exp, sum (16 threads per row) ----
  const int row = tid >> 4;
  const int l16 = tid & 15;
  {
    float m = -3.4e38f;
    for (int j = l16; j < S; j += 16) m = fmaxf(m, sS[row * SP + j]);
    sRed[tid] = m;
  }
  __syncthreads();
  if (tid < 16) {
    float m = sRed[tid * 16];
#pragma unroll
    for (int c = 1; c < 16; ++c) m = fmaxf(m, sRed[tid * 16 + c]);
    sRow[tid] = m;
  }
  __syncthreads();
  {
    const float mrow = sRow[row];
    float sum = 0.f;
    for (int j = l16; j < S; j += 16) {
      float e = __expf(sS[row * SP + j] - mrow);
      sS[row * SP + j] = e;
      sum += e;
    }
    sRed[tid] = sum;
  }
  __syncthreads();
  if (tid < 16) {
    float s = 0.f;
#pragma unroll
    for (int c = 0; c < 16; ++c) s += sRed[tid * 16 + c];
    sRow[16 + tid] = 1.0f / s;
  }
  __syncthreads();

  // ---- stream normalized attn to HBM (dominant memory cost; coalesced) ----
  {
    float* aBase = attnOut + (((size_t)(b * H + h) * S + i0) * S);
    for (int idx = tid; idx < 16 * S; idx += 256) {
      const int rr = idx >> 11;
      const int j  = idx & (S - 1);
      aBase[(size_t)rr * S + j] = sS[rr * SP + j] * sRow[16 + rr];
    }
  }

  // ---- O = P @ V_head : rebuild f16 P fragments from LDS (1/sum folded in) ----
  {
    const float invm = sRow[16 + mr];                 // A-lane row is mr -> uniform factor
    v8f o0 = {}, o1 = {}, o2 = {}, o3 = {};
    const _Float16* vBase = VT + ((size_t)(b * H + h) * DK) * S;
    for (int t = wv; t < S / 32; t += 8) {
      const int j0 = t * 32;
      v16h pa;
#pragma unroll
      for (int e = 0; e < 16; ++e) {
        const int jj = j0 + kb16 + e;
        pa[e] = (_Float16)(sS[mr * SP + jj] * invm);
      }
      o0 = __builtin_amdgcn_wmma_f32_16x16x32_f16(false, pa, false,
             load_b_frag(vBase + (size_t)( 0 + mr) * S + j0 + kb16), (short)0, o0, false, false);
      o1 = __builtin_amdgcn_wmma_f32_16x16x32_f16(false, pa, false,
             load_b_frag(vBase + (size_t)(16 + mr) * S + j0 + kb16), (short)0, o1, false, false);
      o2 = __builtin_amdgcn_wmma_f32_16x16x32_f16(false, pa, false,
             load_b_frag(vBase + (size_t)(32 + mr) * S + j0 + kb16), (short)0, o2, false, false);
      o3 = __builtin_amdgcn_wmma_f32_16x16x32_f16(false, pa, false,
             load_b_frag(vBase + (size_t)(48 + mr) * S + j0 + kb16), (short)0, o3, false, false);
    }
    float* op = sOp + wv * (16 * DK);                 // deterministic per-wave slot
#pragma unroll
    for (int r = 0; r < 8; ++r) {
      op[(il + r) * DK +  0 + mr] = o0[r];
      op[(il + r) * DK + 16 + mr] = o1[r];
      op[(il + r) * DK + 32 + mr] = o2[r];
      op[(il + r) * DK + 48 + mr] = o3[r];
    }
  }
  __syncthreads();

  // ---- deterministic cross-wave reduce, store attn-output rows (f16) ----
  for (int idx = tid; idx < 16 * DK; idx += 256) {
    float s = 0.f;
#pragma unroll
    for (int w = 0; w < 8; ++w) s += sOp[w * (16 * DK) + idx];
    const int rr = idx >> 6;
    const int d  = idx & (DK - 1);
    AO[((size_t)b * S + i0 + rr) * D + h * DK + d] = (_Float16)s;
  }
}

// ---------------------------------------------------------------------------
__global__ void layernorm_kernel(const float* __restrict__ X,
                                 const float* __restrict__ gamma,
                                 const float* __restrict__ beta,
                                 float* __restrict__ out) {
  __shared__ float rs[256], rq[256];
  const int row = blockIdx.x;
  const int tid = threadIdx.x;
  const float* x = X + (size_t)row * D;
  float s = 0.f, q = 0.f;
  for (int j = tid; j < D; j += 256) { float v = x[j]; s += v; q += v * v; }
  rs[tid] = s; rq[tid] = q;
  __syncthreads();
  for (int off = 128; off > 0; off >>= 1) {
    if (tid < off) { rs[tid] += rs[tid + off]; rq[tid] += rq[tid + off]; }
    __syncthreads();
  }
  const float mu  = rs[0] * (1.0f / D);
  const float var = rq[0] * (1.0f / D) - mu * mu;     // matches jnp.var
  const float inv = rsqrtf(var + 1e-6f);
  for (int j = tid; j < D; j += 256)
    out[(size_t)row * D + j] = (x[j] - mu) * inv * gamma[j] + beta[j];
}

// ---------------------------------------------------------------------------
extern "C" void kernel_launch(void* const* d_in, const int* in_sizes, int n_in,
                              void* d_out, int out_size, void* d_ws, size_t ws_size,
                              hipStream_t stream) {
  const float* queries = (const float*)d_in[0];
  const float* keys    = (const float*)d_in[1];
  const float* values  = (const float*)d_in[2];
  const int*   mask    = (const int*)d_in[3];
  const float* Wq      = (const float*)d_in[4];
  const float* Wk      = (const float*)d_in[5];
  const float* Wv      = (const float*)d_in[6];
  const float* Wfc     = (const float*)d_in[7];
  const float* gamma   = (const float*)d_in[8];
  const float* beta    = (const float*)d_in[9];

  const size_t E  = (size_t)Bq * S * D;     // 4,194,304 activation elements
  const size_t W2 = (size_t)D * D * sizeof(_Float16);

  char* p = (char*)d_ws;
  _Float16* xq   = (_Float16*)p; p += E * 2;
  _Float16* xk   = (_Float16*)p; p += E * 2;
  _Float16* xv   = (_Float16*)p; p += E * 2;
  _Float16* WqT  = (_Float16*)p; p += W2;
  _Float16* WkT  = (_Float16*)p; p += W2;
  _Float16* WvT  = (_Float16*)p; p += W2;
  _Float16* WfcT = (_Float16*)p; p += W2;
  _Float16* qh   = (_Float16*)p; p += E * 2;
  _Float16* kh   = (_Float16*)p; p += E * 2;
  _Float16* vh   = (_Float16*)p; p += E * 2;
  _Float16* vT   = (_Float16*)p; p += E * 2;
  _Float16* ao   = xq;           // reuse: f16 inputs consumed after projections
  float*    fc   = (float*)xk;   // reuse xk+xv (16 MB) for f32 fc output

  float* outF    = (float*)d_out;
  float* attnOut = outF + E;     // tuple output #2: (B,H,S,S) f32

  dim3 blk(256);
  cvt_f32_to_f16<<<(int)(E / 256), blk, 0, stream>>>(queries, xq, (int)E);
  cvt_f32_to_f16<<<(int)(E / 256), blk, 0, stream>>>(keys,    xk, (int)E);
  cvt_f32_to_f16<<<(int)(E / 256), blk, 0, stream>>>(values,  xv, (int)E);
  wtrans_f16<<<(D * D) / 256, blk, 0, stream>>>(Wq,  WqT);
  wtrans_f16<<<(D * D) / 256, blk, 0, stream>>>(Wk,  WkT);
  wtrans_f16<<<(D * D) / 256, blk, 0, stream>>>(Wv,  WvT);
  wtrans_f16<<<(D * D) / 256, blk, 0, stream>>>(Wfc, WfcT);

  const int M = Bq * S;                                    // 4096 rows
  const int gemmBlocks = (M / 16) * (D / 64) * 32 / 256;   // 512 blocks of 8 waves
  gemm_wmma_f16<<<gemmBlocks, blk, 0, stream>>>(xq, WqT, qh, nullptr, nullptr, M, D, D, D, D, D);
  gemm_wmma_f16<<<gemmBlocks, blk, 0, stream>>>(xk, WkT, kh, nullptr, nullptr, M, D, D, D, D, D);
  gemm_wmma_f16<<<gemmBlocks, blk, 0, stream>>>(xv, WvT, vh, nullptr, nullptr, M, D, D, D, D, D);

  vtrans_f16<<<(int)(E / 256), blk, 0, stream>>>(vh, vT);

  const size_t smem = (size_t)(16 * SP + 256 + 32 + 8 * 16 * DK) * sizeof(float); // ~165 KB < 320 KB WGP LDS
  attn_kernel<<<Bq * H * (S / 16), blk, smem, stream>>>(qh, kh, vT, mask, attnOut, ao);

  gemm_wmma_f16<<<gemmBlocks, blk, 0, stream>>>(ao, WfcT, nullptr, fc, queries, M, D, D, D, D, D);

  layernorm_kernel<<<M, blk, 0, stream>>>(fc, gamma, beta, outF);
}
